// GptOssDecoderLayer_19095424598732
// MI455X (gfx1250) — matrix operand order repaired and checked by reading, not compiled, not run on hardware
//
#include <hip/hip_runtime.h>
#include <stdint.h>

typedef __attribute__((ext_vector_type(16))) _Float16 v16h;
typedef __attribute__((ext_vector_type(2)))  __fp16   v2hf;   // cvt_pkrtz result type
typedef __attribute__((ext_vector_type(8)))  float    v8f;

#define S_TOK 2048
#define HID   1024
#define NHEAD 16
#define NKVH  4
#define HDIM  64
#define NEXP  8
#define FFD   1024
#define QKVN  1152
#define LDST  40      // LDS row stride in halfs (80B = 5x16B, keeps b128 rows aligned)

#define WMMA32(a,b,c) __builtin_amdgcn_wmma_f32_16x16x32_f16(false,(a),false,(b),(short)0,(c),false,false)

__device__ __forceinline__ v8f zero8() {
  v8f z = {0.f,0.f,0.f,0.f,0.f,0.f,0.f,0.f};
  return z;
}

__device__ __forceinline__ uint32_t lds_off(const void* p) {
  // flat LDS aperture: addr[31:0] is the wave-relative LDS byte offset
  return (uint32_t)(uintptr_t)p;
}

// async per-lane 16B copy global -> LDS, tracked by ASYNCcnt
__device__ __forceinline__ void async_b128(uint32_t lds, uint64_t gaddr) {
  asm volatile("global_load_async_to_lds_b128 %0, %1, off"
               :: "v"(lds), "v"(gaddr) : "memory");
}
__device__ __forceinline__ void wait_async0() {
  asm volatile("s_wait_asynccnt 0x0" ::: "memory");
}

// pack two f32 -> packed f16 pair and store as one 32-bit LDS write
__device__ __forceinline__ void pack_store(_Float16* dst, float x, float y) {
  v2hf pk = __builtin_amdgcn_cvt_pkrtz(x, y);
  *reinterpret_cast<v2hf*>(dst) = pk;
}

// A fragment (16x32 f16): lane half splits K 0-15 / 16-31 (matches fragB)
__device__ __forceinline__ v16h fragA(const _Float16* base, int stride, int lane) {
  const int m = lane & 15, hf = lane >> 4;
  v16h a;
  const _Float16* row = base + m * stride;
#pragma unroll
  for (int v = 0; v < 8; ++v) {
    int k = 2 * v + (hf ? 16 : 0);
    a[2 * v]     = row[k];
    a[2 * v + 1] = row[k + 1];
  }
  return a;
}

// B fragment (32x16 f16): lane = output column n; K split 0-15 / 16-31 across halves.
__device__ __forceinline__ v16h fragB(const _Float16* base, int stride, int lane) {
  const int n = lane & 15, hf = lane >> 4;
  v16h b;
  const _Float16* row = base + n * stride;
#pragma unroll
  for (int v = 0; v < 8; ++v) {
    int k = 2 * v + (hf ? 16 : 0);
    b[2 * v]     = row[k];
    b[2 * v + 1] = row[k + 1];
  }
  return b;
}

__device__ __forceinline__ float redmax16(float v) {
#pragma unroll
  for (int m = 8; m >= 1; m >>= 1) v = fmaxf(v, __shfl_xor(v, m, 32));
  return v;
}
__device__ __forceinline__ float redsum16(float v) {
#pragma unroll
  for (int m = 8; m >= 1; m >>= 1) v += __shfl_xor(v, m, 32);
  return v;
}

// ---------------------------------------------------------------------------
// res1 = x (+ r); h = rmsnorm(res1) * w   (r / res_out optional)
// ---------------------------------------------------------------------------
__global__ void __launch_bounds__(256)
add_rmsnorm_kernel(const float* __restrict__ x, const float* __restrict__ r,
                   const float* __restrict__ w, float* __restrict__ res_out,
                   _Float16* __restrict__ hf) {
  const int t = blockIdx.x;
  const int tid = threadIdx.x, wave = tid >> 5, lane = tid & 31;
  __shared__ float red[8];
  __shared__ float rinv;
  const float* xa = x + (size_t)t * HID;
  float vals[4];
  float ss = 0.f;
#pragma unroll
  for (int i = 0; i < 4; ++i) {
    int idx = tid + i * 256;
    float v = xa[idx] + (r ? r[(size_t)t * HID + idx] : 0.f);
    vals[i] = v;
    ss += v * v;
  }
#pragma unroll
  for (int m = 16; m >= 1; m >>= 1) ss += __shfl_xor(ss, m, 32);
  if (lane == 0) red[wave] = ss;
  __syncthreads();
  if (tid == 0) {
    float s = 0.f;
#pragma unroll
    for (int i = 0; i < 8; ++i) s += red[i];
    rinv = rsqrtf(s * (1.f / HID) + 1e-5f);
  }
  __syncthreads();
  const float ri = rinv;
#pragma unroll
  for (int i = 0; i < 4; ++i) {
    int idx = tid + i * 256;
    float v = vals[i];
    if (res_out) res_out[(size_t)t * HID + idx] = v;
    hf[(size_t)t * HID + idx] = (_Float16)(v * ri * w[idx]);
  }
}

// ---------------------------------------------------------------------------
// Generic GEMM: C[M,N] = A_f16[M,K] * W_f32[N,K]^T + bias (+ addend), f32 out.
// 64x64 block tile, 4 waves x 32x32 WMMA, double-buffered LDS:
//   A tile via global_load_async_to_lds_b128, W tile reg-staged + pk_rtz pack.
// ---------------------------------------------------------------------------
template <bool HAS_ADD>
__global__ void __launch_bounds__(128)
gemm_f16w32_kernel(const _Float16* __restrict__ A, const float* __restrict__ W,
                   const float* __restrict__ bias, const float* __restrict__ addend,
                   float* __restrict__ C, int N, int K) {
  __shared__ _Float16 As[2][64 * LDST];
  __shared__ _Float16 Ws[2][64 * LDST];
  const int m0 = blockIdx.x * 64, n0 = blockIdx.y * 64;
  const int tid = threadIdx.x, wave = tid >> 5, lane = tid & 31;
  const int wm = (wave >> 1) * 32, wn = (wave & 1) * 32;

  auto loadA_async = [&](int k0, int buf) {
#pragma unroll
    for (int i = 0; i < 2; ++i) {            // 64x32 halfs = 256 x b128
      int u = tid + i * 128, rr = u >> 2, c = (u & 3) * 8;
      async_b128(lds_off(&As[buf][rr * LDST + c]),
                 (uint64_t)(uintptr_t)(A + (size_t)(m0 + rr) * K + k0 + c));
    }
  };
  auto loadW_regs = [&](int k0, float2* wr) {
#pragma unroll
    for (int i = 0; i < 8; ++i) {            // 64x32 f32 as 1024 float2
      int u = tid + i * 128, rr = u >> 4, c2 = (u & 15) * 2;
      wr[i] = *reinterpret_cast<const float2*>(W + (size_t)(n0 + rr) * K + k0 + c2);
    }
  };
  auto storeW_lds = [&](const float2* wr, int buf) {
#pragma unroll
    for (int i = 0; i < 8; ++i) {
      int u = tid + i * 128, rr = u >> 4, c2 = (u & 15) * 2;
      pack_store(&Ws[buf][rr * LDST + c2], wr[i].x, wr[i].y);
    }
  };

  v8f acc[2][2];
  acc[0][0] = acc[0][1] = acc[1][0] = acc[1][1] = zero8();

  float2 wreg[8];
  loadA_async(0, 0);
  loadW_regs(0, wreg);
  storeW_lds(wreg, 0);
  wait_async0();
  __syncthreads();

  int buf = 0;
  for (int k0 = 0; k0 < K; k0 += 32) {
    const bool has_next = (k0 + 32) < K;
    if (has_next) {
      loadA_async(k0 + 32, buf ^ 1);         // overlap with WMMA below
      loadW_regs(k0 + 32, wreg);
    }
    v16h a0 = fragA(As[buf] + wm * LDST, LDST, lane);
    v16h a1 = fragA(As[buf] + (wm + 16) * LDST, LDST, lane);
    v16h b0 = fragB(Ws[buf] + wn * LDST, LDST, lane);
    v16h b1 = fragB(Ws[buf] + (wn + 16) * LDST, LDST, lane);
    acc[0][0] = WMMA32(a0, b0, acc[0][0]);
    acc[0][1] = WMMA32(a0, b1, acc[0][1]);
    acc[1][0] = WMMA32(a1, b0, acc[1][0]);
    acc[1][1] = WMMA32(a1, b1, acc[1][1]);
    if (has_next) {
      storeW_lds(wreg, buf ^ 1);
      wait_async0();
    }
    __syncthreads();
    buf ^= 1;
  }

  const int ln = lane & 15, lh = lane >> 4;
#pragma unroll
  for (int i = 0; i < 2; ++i)
#pragma unroll
    for (int j = 0; j < 2; ++j) {
      int nn = n0 + wn + j * 16 + ln;
      int mb = m0 + wm + i * 16 + lh * 8;
      float bv = bias[nn];
      float* cp = C + (size_t)mb * N + nn;
      const float* ap = HAS_ADD ? (addend + (size_t)mb * N + nn) : nullptr;
#pragma unroll
      for (int v = 0; v < 8; ++v) {
        float val = acc[i][j][v] + bv;
        if (HAS_ADD) val += ap[(size_t)v * N];
        cp[(size_t)v * N] = val;
      }
    }
}

// ---------------------------------------------------------------------------
// RoPE + layout shuffle.  Q scaled by D^-0.5 here.  V stored transposed [D][S].
// ---------------------------------------------------------------------------
__global__ void __launch_bounds__(256)
rope_kernel(const float* __restrict__ qkv, const int* __restrict__ pos,
            _Float16* __restrict__ Qf, _Float16* __restrict__ Kf,
            _Float16* __restrict__ Vt) {
  const int t = blockIdx.x;
  const float p = (float)pos[t];
  const float* row = qkv + (size_t)t * QKVN;
  for (int i = threadIdx.x; i < NHEAD * HDIM; i += 256) {
    int hh = i >> 6, d = i & 63, dh = d & 31;
    float ang = p * __powf(10000.f, -(float)dh * (1.f / 32.f));
    float sn, cs;
    __sincosf(ang, &sn, &cs);
    float xv = row[i];
    float rot = (d < 32) ? -row[(hh << 6) + d + 32] : row[(hh << 6) + d - 32];
    Qf[((size_t)hh * S_TOK + t) * HDIM + d] = (_Float16)((xv * cs + rot * sn) * 0.125f);
  }
  for (int i = threadIdx.x; i < NKVH * HDIM; i += 256) {
    int hh = i >> 6, d = i & 63, dh = d & 31;
    float ang = p * __powf(10000.f, -(float)dh * (1.f / 32.f));
    float sn, cs;
    __sincosf(ang, &sn, &cs);
    float xv = row[1024 + i];
    float rot = (d < 32) ? -row[1024 + (hh << 6) + d + 32] : row[1024 + (hh << 6) + d - 32];
    Kf[((size_t)hh * S_TOK + t) * HDIM + d] = (_Float16)(xv * cs + rot * sn);
  }
  for (int i = threadIdx.x; i < NKVH * HDIM; i += 256) {
    int hh = i >> 6, d = i & 63;
    Vt[((size_t)hh * HDIM + d) * S_TOK + t] = (_Float16)row[1280 + i];
  }
}

// ---------------------------------------------------------------------------
// Flash attention with sink, GQA 4:1.  One wave per 16-query tile.
// ---------------------------------------------------------------------------
__global__ void __launch_bounds__(128)
attn_kernel(const _Float16* __restrict__ Qf, const _Float16* __restrict__ Kf,
            const _Float16* __restrict__ Vt, const float* __restrict__ sinks,
            _Float16* __restrict__ attnOut) {
  __shared__ _Float16 Pl[4][16 * 34];
  const int h = blockIdx.x;
  const int wave = threadIdx.x >> 5, lane = threadIdx.x & 31;
  const int q0 = blockIdx.y * 64 + wave * 16;
  const int kvh = h >> 2;
  const _Float16* Q  = Qf + ((size_t)h * S_TOK + q0) * HDIM;
  const _Float16* Kp = Kf + (size_t)kvh * S_TOK * HDIM;
  const _Float16* Vp = Vt + (size_t)kvh * HDIM * S_TOK;
  const int ln = lane & 15, lh = lane >> 4;

  v16h aq[2];                          // Q fragment, loop-invariant, K-dim = 64
#pragma unroll
  for (int ks = 0; ks < 2; ++ks) {
#pragma unroll
    for (int v = 0; v < 8; ++v) {
      int k = ks * 32 + 2 * v + (lh ? 16 : 0);
      aq[ks][2 * v]     = Q[(size_t)ln * HDIM + k];
      aq[ks][2 * v + 1] = Q[(size_t)ln * HDIM + k + 1];
    }
  }
  float mrow[8], lrow[8];
#pragma unroll
  for (int v = 0; v < 8; ++v) { mrow[v] = -1e30f; lrow[v] = 0.f; }
  v8f acc[4];
  acc[0] = acc[1] = acc[2] = acc[3] = zero8();
  _Float16* P = Pl[wave];
  const int qmax = q0 + 15;

  for (int k0 = 0; k0 <= qmax; k0 += 32) {
    v8f s0 = zero8(), s1 = zero8();
#pragma unroll
    for (int ks = 0; ks < 2; ++ks) {
      v16h b0, b1;
#pragma unroll
      for (int v = 0; v < 8; ++v) {
        int d = ks * 32 + 2 * v + (lh ? 16 : 0);
        const _Float16* k0r = Kp + (size_t)(k0 + ln) * HDIM + d;
        const _Float16* k1r = Kp + (size_t)(k0 + 16 + ln) * HDIM + d;
        b0[2 * v] = k0r[0]; b0[2 * v + 1] = k0r[1];
        b1[2 * v] = k1r[0]; b1[2 * v + 1] = k1r[1];
      }
      s0 = WMMA32(aq[ks], b0, s0);
      s1 = WMMA32(aq[ks], b1, s1);
    }
#pragma unroll
    for (int v = 0; v < 8; ++v) {
      int qrow = q0 + lh * 8 + v;
      float v0 = ((k0 + ln) <= qrow) ? s0[v] : -1e30f;
      float v1 = ((k0 + 16 + ln) <= qrow) ? s1[v] : -1e30f;
      float cm = redmax16(fmaxf(v0, v1));
      float mnew = fmaxf(mrow[v], cm);
      float scale = __expf(mrow[v] - mnew);
      float p0 = __expf(v0 - mnew);
      float p1 = __expf(v1 - mnew);
      lrow[v] = lrow[v] * scale + redsum16(p0 + p1);
      mrow[v] = mnew;
      acc[0][v] *= scale; acc[1][v] *= scale; acc[2][v] *= scale; acc[3][v] *= scale;
      int mr = lh * 8 + v;
      P[mr * 34 + ln]      = (_Float16)p0;
      P[mr * 34 + 16 + ln] = (_Float16)p1;
    }
    v16h ap = fragA(P, 34, lane);      // wave-local LDS round-trip (dscnt-guarded)
#pragma unroll
    for (int dt = 0; dt < 4; ++dt) {
      v16h b;
#pragma unroll
      for (int v = 0; v < 8; ++v) {
        int kk = 2 * v + (lh ? 16 : 0);
        const _Float16* vr = Vp + (size_t)(dt * 16 + ln) * S_TOK + k0 + kk;
        b[2 * v] = vr[0]; b[2 * v + 1] = vr[1];
      }
      acc[dt] = WMMA32(ap, b, acc[dt]);
    }
  }
  const float sk = sinks[h];
#pragma unroll
  for (int v = 0; v < 8; ++v) {
    float inv = 1.f / (lrow[v] + __expf(sk - mrow[v]));   // attention sink
    int t = q0 + lh * 8 + v;
#pragma unroll
    for (int dt = 0; dt < 4; ++dt)
      attnOut[(size_t)t * (NHEAD * HDIM) + h * HDIM + dt * 16 + ln] =
          (_Float16)(acc[dt][v] * inv);
  }
}

// ---------------------------------------------------------------------------
// Router: per-token 8 logits (one wave each), top-2 softmax, atomic compaction.
// ---------------------------------------------------------------------------
__global__ void __launch_bounds__(256)
router_kernel(const _Float16* __restrict__ h2, const float* __restrict__ rw,
              const float* __restrict__ rb, int* __restrict__ cnt,
              int* __restrict__ tok, float* __restrict__ wts) {
  const int t = blockIdx.x;
  const int wave = threadIdx.x >> 5, lane = threadIdx.x & 31;
  __shared__ float lg[NEXP];
  const _Float16* x = h2 + (size_t)t * HID;
  const float* we = rw + (size_t)wave * HID;
  float s = 0.f;
  for (int j = lane; j < HID; j += 32) s += (float)x[j] * we[j];
#pragma unroll
  for (int m = 16; m >= 1; m >>= 1) s += __shfl_xor(s, m, 32);
  if (lane == 0) lg[wave] = s + rb[wave];
  __syncthreads();
  if (threadIdx.x == 0) {
    int i0 = 0;
#pragma unroll
    for (int e = 1; e < NEXP; ++e) if (lg[e] > lg[i0]) i0 = e;
    int i1 = (i0 == 0) ? 1 : 0;
#pragma unroll
    for (int e = 0; e < NEXP; ++e) if (e != i0 && lg[e] > lg[i1]) i1 = e;
    float mx = fmaxf(lg[i0], lg[i1]);
    float e0 = __expf(lg[i0] - mx), e1 = __expf(lg[i1] - mx);
    float inv = 1.f / (e0 + e1);
    int s0 = atomicAdd(&cnt[i0], 1);
    tok[i0 * S_TOK + s0] = t; wts[i0 * S_TOK + s0] = e0 * inv;
    int s1 = atomicAdd(&cnt[i1], 1);
    tok[i1 * S_TOK + s1] = t; wts[i1 * S_TOK + s1] = e1 * inv;
  }
}

// ---------------------------------------------------------------------------
// MoE gate_up GEMM (gathered rows, async A) + fused clamp/SiLU -> act f16.
// Rows >= count alias slot 0 (their accumulator rows are never stored).
// ---------------------------------------------------------------------------
__global__ void __launch_bounds__(128)
moe_gateup_kernel(const _Float16* __restrict__ h2, const float* __restrict__ Wall,
                  const float* __restrict__ Ball, const int* __restrict__ cnt,
                  const int* __restrict__ tok, _Float16* __restrict__ act) {
  __shared__ _Float16 As[2][64 * LDST];
  __shared__ _Float16 Ws[2][64 * LDST];
  __shared__ int rowTok[64];
  const int e = blockIdx.z;
  const int count = cnt[e];
  const int m0 = blockIdx.x * 64;
  if (m0 >= count) return;
  const int n0 = blockIdx.y * 64;
  const float* W = Wall + (size_t)e * (2 * FFD) * HID;
  const float* B = Ball + (size_t)e * (2 * FFD);
  const int* tk = tok + (size_t)e * S_TOK;
  const int tid = threadIdx.x, wave = tid >> 5, lane = tid & 31;
  const int wm = (wave >> 1) * 32, wn = (wave & 1) * 32;
  if (tid < 64) { int mm = m0 + tid; rowTok[tid] = tk[(mm < count) ? mm : 0]; }
  __syncthreads();

  auto loadA_async = [&](int k0, int buf) {
#pragma unroll
    for (int i = 0; i < 2; ++i) {
      int u = tid + i * 128, rr = u >> 2, c = (u & 3) * 8;
      async_b128(lds_off(&As[buf][rr * LDST + c]),
                 (uint64_t)(uintptr_t)(h2 + (size_t)rowTok[rr] * HID + k0 + c));
    }
  };
  auto loadW_regs = [&](int k0, float2* wr) {
#pragma unroll
    for (int i = 0; i < 8; ++i) {
      int u = tid + i * 128, rr = u >> 4, c2 = (u & 15) * 2;
      wr[i] = *reinterpret_cast<const float2*>(W + (size_t)(n0 + rr) * HID + k0 + c2);
    }
  };
  auto storeW_lds = [&](const float2* wr, int buf) {
#pragma unroll
    for (int i = 0; i < 8; ++i) {
      int u = tid + i * 128, rr = u >> 4, c2 = (u & 15) * 2;
      pack_store(&Ws[buf][rr * LDST + c2], wr[i].x, wr[i].y);
    }
  };

  v8f acc[2][2];
  acc[0][0] = acc[0][1] = acc[1][0] = acc[1][1] = zero8();
  float2 wreg[8];
  loadA_async(0, 0);
  loadW_regs(0, wreg);
  storeW_lds(wreg, 0);
  wait_async0();
  __syncthreads();

  int buf = 0;
  for (int k0 = 0; k0 < HID; k0 += 32) {
    const bool has_next = (k0 + 32) < HID;
    if (has_next) { loadA_async(k0 + 32, buf ^ 1); loadW_regs(k0 + 32, wreg); }
    v16h a0 = fragA(As[buf] + wm * LDST, LDST, lane);
    v16h a1 = fragA(As[buf] + (wm + 16) * LDST, LDST, lane);
    v16h b0 = fragB(Ws[buf] + wn * LDST, LDST, lane);
    v16h b1 = fragB(Ws[buf] + (wn + 16) * LDST, LDST, lane);
    acc[0][0] = WMMA32(a0, b0, acc[0][0]);
    acc[0][1] = WMMA32(a0, b1, acc[0][1]);
    acc[1][0] = WMMA32(a1, b0, acc[1][0]);
    acc[1][1] = WMMA32(a1, b1, acc[1][1]);
    if (has_next) { storeW_lds(wreg, buf ^ 1); wait_async0(); }
    __syncthreads();
    buf ^= 1;
  }

  const int ln = lane & 15, lh = lane >> 4;
#pragma unroll
  for (int i = 0; i < 2; ++i)
#pragma unroll
    for (int j = 0; j < 2; ++j) {
      int n = n0 + wn + j * 16 + ln;       // even n = gate col, odd n = up col
      float bv = B[n];
      int mbase = m0 + wm + i * 16 + lh * 8;
#pragma unroll
      for (int v = 0; v < 8; ++v) {
        float val = acc[i][j][v] + bv;
        float partner = __shfl_xor(val, 1, 32);   // pair gate<->up columns
        if ((ln & 1) == 0) {
          float g = fminf(val, 7.0f);
          float u = fminf(fmaxf(partner, -7.0f), 7.0f);
          float a = (u + 1.f) * g * (1.f / (1.f + __expf(-1.702f * g)));
          int slot = mbase + v;
          if (slot < count)
            act[((size_t)e * S_TOK + slot) * FFD + (n >> 1)] = (_Float16)a;
        }
      }
    }
}

// ---------------------------------------------------------------------------
// MoE down GEMM over compacted slots (async A); scatter w*(act.W^T + b).
// ---------------------------------------------------------------------------
__global__ void __launch_bounds__(128)
moe_down_kernel(const _Float16* __restrict__ act, const float* __restrict__ Wall,
                const float* __restrict__ Ball, const int* __restrict__ cnt,
                const int* __restrict__ tok, const float* __restrict__ wts,
                float* __restrict__ moe) {
  __shared__ _Float16 As[2][64 * LDST];
  __shared__ _Float16 Ws[2][64 * LDST];
  const int e = blockIdx.z;
  const int count = cnt[e];
  const int m0 = blockIdx.x * 64;
  if (m0 >= count) return;
  const int n0 = blockIdx.y * 64;
  const _Float16* A = act + (size_t)e * S_TOK * FFD;
  const float* W = Wall + (size_t)e * HID * FFD;
  const float* B = Ball + (size_t)e * HID;
  const int tid = threadIdx.x, wave = tid >> 5, lane = tid & 31;
  const int wm = (wave >> 1) * 32, wn = (wave & 1) * 32;

  auto loadA_async = [&](int k0, int buf) {
#pragma unroll
    for (int i = 0; i < 2; ++i) {
      int u = tid + i * 128, rr = u >> 2, c = (u & 3) * 8;
      int sl = (m0 + rr < count) ? (m0 + rr) : 0;
      async_b128(lds_off(&As[buf][rr * LDST + c]),
                 (uint64_t)(uintptr_t)(A + (size_t)sl * FFD + k0 + c));
    }
  };
  auto loadW_regs = [&](int k0, float2* wr) {
#pragma unroll
    for (int i = 0; i < 8; ++i) {
      int u = tid + i * 128, rr = u >> 4, c2 = (u & 15) * 2;
      wr[i] = *reinterpret_cast<const float2*>(W + (size_t)(n0 + rr) * FFD + k0 + c2);
    }
  };
  auto storeW_lds = [&](const float2* wr, int buf) {
#pragma unroll
    for (int i = 0; i < 8; ++i) {
      int u = tid + i * 128, rr = u >> 4, c2 = (u & 15) * 2;
      pack_store(&Ws[buf][rr * LDST + c2], wr[i].x, wr[i].y);
    }
  };

  v8f acc[2][2];
  acc[0][0] = acc[0][1] = acc[1][0] = acc[1][1] = zero8();
  float2 wreg[8];
  loadA_async(0, 0);
  loadW_regs(0, wreg);
  storeW_lds(wreg, 0);
  wait_async0();
  __syncthreads();

  int buf = 0;
  for (int k0 = 0; k0 < FFD; k0 += 32) {
    const bool has_next = (k0 + 32) < FFD;
    if (has_next) { loadA_async(k0 + 32, buf ^ 1); loadW_regs(k0 + 32, wreg); }
    v16h a0 = fragA(As[buf] + wm * LDST, LDST, lane);
    v16h a1 = fragA(As[buf] + (wm + 16) * LDST, LDST, lane);
    v16h b0 = fragB(Ws[buf] + wn * LDST, LDST, lane);
    v16h b1 = fragB(Ws[buf] + (wn + 16) * LDST, LDST, lane);
    acc[0][0] = WMMA32(a0, b0, acc[0][0]);
    acc[0][1] = WMMA32(a0, b1, acc[0][1]);
    acc[1][0] = WMMA32(a1, b0, acc[1][0]);
    acc[1][1] = WMMA32(a1, b1, acc[1][1]);
    if (has_next) { storeW_lds(wreg, buf ^ 1); wait_async0(); }
    __syncthreads();
    buf ^= 1;
  }

  const int ln = lane & 15, lh = lane >> 4;
#pragma unroll
  for (int i = 0; i < 2; ++i)
#pragma unroll
    for (int j = 0; j < 2; ++j) {
      int n = n0 + wn + j * 16 + ln;
      float bv = B[n];
      int mbase = m0 + wm + i * 16 + lh * 8;
#pragma unroll
      for (int v = 0; v < 8; ++v) {
        int slot = mbase + v;
        if (slot < count) {
          int t = tok[e * S_TOK + slot];
          float wgt = wts[e * S_TOK + slot];
          atomicAdd(&moe[(size_t)t * HID + n], wgt * (acc[i][j][v] + bv));
        }
      }
    }
}

// ---------------------------------------------------------------------------
extern "C" void kernel_launch(void* const* d_in, const int* in_sizes, int n_in,
                              void* d_out, int out_size, void* d_ws, size_t ws_size,
                              hipStream_t stream) {
  (void)in_sizes; (void)n_in; (void)out_size; (void)ws_size;
  const float* hs    = (const float*)d_in[0];
  const float* resid = (const float*)d_in[1];
  const int*   pos   = (const int*)d_in[2];
  const float* ln1   = (const float*)d_in[3];
  const float* ln2   = (const float*)d_in[4];
  const float* qkvW  = (const float*)d_in[5];
  const float* qkvB  = (const float*)d_in[6];
  const float* sinks = (const float*)d_in[7];
  const float* oW    = (const float*)d_in[8];
  const float* oB    = (const float*)d_in[9];
  const float* rW    = (const float*)d_in[10];
  const float* rB    = (const float*)d_in[11];
  const float* guW   = (const float*)d_in[12];
  const float* guB   = (const float*)d_in[13];
  const float* dnW   = (const float*)d_in[14];
  const float* dnB   = (const float*)d_in[15];

  float* moe  = (float*)d_out;                    // output 0
  float* res2 = moe + (size_t)S_TOK * HID;        // output 1

  char* base = (char*)d_ws;
  size_t off = 0;
  auto take = [&](size_t bytes) -> void* {
    void* p = base + off;
    off += (bytes + 255) & ~(size_t)255;
    return p;
  };
  float*    res1 = (float*)   take((size_t)S_TOK * HID * 4);
  _Float16* h1f  = (_Float16*)take((size_t)S_TOK * HID * 2);
  float*    qkv  = (float*)   take((size_t)S_TOK * QKVN * 4);
  _Float16* Qf   = (_Float16*)take((size_t)NHEAD * S_TOK * HDIM * 2);
  _Float16* Kf   = (_Float16*)take((size_t)NKVH * S_TOK * HDIM * 2);
  _Float16* Vt   = (_Float16*)take((size_t)NKVH * HDIM * S_TOK * 2);
  _Float16* attn = (_Float16*)take((size_t)S_TOK * NHEAD * HDIM * 2);
  _Float16* h2f  = (_Float16*)take((size_t)S_TOK * HID * 2);
  int*      cnt  = (int*)     take((size_t)NEXP * 4);
  int*      tokl = (int*)     take((size_t)NEXP * S_TOK * 4);
  float*    wtl  = (float*)   take((size_t)NEXP * S_TOK * 4);
  _Float16* actb = (_Float16*)take((size_t)NEXP * S_TOK * FFD * 2);

  (void)hipMemsetAsync(moe, 0, (size_t)S_TOK * HID * 4, stream);
  (void)hipMemsetAsync(cnt, 0, NEXP * 4, stream);

  add_rmsnorm_kernel<<<S_TOK, 256, 0, stream>>>(hs, resid, ln1, res1, h1f);
  gemm_f16w32_kernel<false><<<dim3(S_TOK / 64, QKVN / 64), 128, 0, stream>>>(
      h1f, qkvW, qkvB, nullptr, qkv, QKVN, HID);
  rope_kernel<<<S_TOK, 256, 0, stream>>>(qkv, pos, Qf, Kf, Vt);
  attn_kernel<<<dim3(NHEAD, S_TOK / 64), 128, 0, stream>>>(Qf, Kf, Vt, sinks, attn);
  gemm_f16w32_kernel<true><<<dim3(S_TOK / 64, HID / 64), 128, 0, stream>>>(
      attn, oW, oB, res1, res2, HID, NHEAD * HDIM);
  add_rmsnorm_kernel<<<S_TOK, 256, 0, stream>>>(res2, nullptr, ln2, nullptr, h2f);
  router_kernel<<<S_TOK, 256, 0, stream>>>(h2f, rW, rB, cnt, tokl, wtl);
  moe_gateup_kernel<<<dim3(S_TOK / 64, (2 * FFD) / 64, NEXP), 128, 0, stream>>>(
      h2f, guW, guB, cnt, tokl, actb);
  moe_down_kernel<<<dim3(S_TOK / 64, HID / 64, NEXP), 128, 0, stream>>>(
      actb, dnW, dnB, cnt, tokl, wtl, moe);
}
